// Attention_13615046328582
// MI455X (gfx1250) — matrix operand order repaired and checked by reading, not compiled, not run on hardware
//
#include <hip/hip_runtime.h>
#include <hip/hip_bf16.h>

typedef unsigned short u16t;
typedef __attribute__((ext_vector_type(16))) __bf16 v16bf;
typedef __attribute__((ext_vector_type(8)))  float  v8f;
typedef __attribute__((ext_vector_type(4)))  unsigned int v4u;
typedef __attribute__((ext_vector_type(8)))  int  v8i;
typedef __attribute__((ext_vector_type(4)))  int  v4i;

#define B_ 4
#define S_ 2048
#define E_ 2048
#define H_ 16
#define D_ 128
#define M_ (B_*S_)   // 8192 rows

__device__ __forceinline__ u16t f2b(float f) {
  unsigned u = __float_as_uint(f);
  u += 0x7fffu + ((u >> 16) & 1u);   // round-to-nearest-even
  return (u16t)(u >> 16);
}
__device__ __forceinline__ float b2f(u16t h) {
  return __uint_as_float(((unsigned)h) << 16);
}

union Frag { v16bf v; uint4 u[2]; };

__device__ __forceinline__ v16bf ldfrag(const u16t* p0, const u16t* p1) {
  Frag f;
  f.u[0] = *reinterpret_cast<const uint4*>(p0);
  f.u[1] = *reinterpret_cast<const uint4*>(p1);
  return f.v;
}

__device__ __forceinline__ v8f wmma_bf16(v16bf a, v16bf b, v8f c) {
  // D = A(16x32 bf16) x B(32x16 bf16) + C(16x16 f32)
  return __builtin_amdgcn_wmma_f32_16x16x32_bf16(
      false, a, false, b, (short)0, c, false, false);
}

// LDS byte address (AS3 ptrtoint) of a __shared__ object
typedef __attribute__((address_space(3))) void lds_void;
__device__ __forceinline__ unsigned lds_off(void* p) {
  return (unsigned)(unsigned long long)(lds_void*)p;
}

// ------------------------------------------------------------------ TDM tile load
// DMA a rows x width (bf16 elements) tile from global (row stride `row_stride`
// elements) into LDS, padding each row per pad codes (interval: 2^(c+1) dwords,
// amount: c+1 dwords).
__device__ __forceinline__ void tdm_load_2d(unsigned lds_byte, const u16t* gtile,
                                            unsigned width, unsigned rows,
                                            unsigned row_stride, unsigned rows_total,
                                            int pad_interval_c, int pad_amount_c) {
  unsigned long long ga = (unsigned long long)gtile;
  v4u g0;
  g0[0] = 1u;                                  // count=1, user descriptor
  g0[1] = lds_byte;                            // lds_addr (bytes)
  g0[2] = (unsigned)ga;                        // global_addr[31:0]
  g0[3] = (unsigned)((ga >> 32) & 0x01FFFFFFu) // global_addr[56:32]
        | (2u << 30);                          // type = 2 ("image")
  v8i g1;
  g1[0] = (1 << 16)                            // data_size = 2 bytes
        | (1 << 20)                            // pad_enable
        | (pad_interval_c << 22)               // pad interval code
        | (pad_amount_c << 25);                // pad amount code
  g1[1] = (int)((row_stride & 0xFFFFu) << 16); // tensor_dim0 lo (bits 79:48)
  g1[2] = (int)((row_stride >> 16) & 0xFFFFu)  // tensor_dim0 hi
        | (int)((rows_total & 0xFFFFu) << 16); // tensor_dim1 lo (bits 111:80)
  g1[3] = (int)((rows_total >> 16) & 0xFFFFu)  // tensor_dim1 hi
        | (int)(width << 16);                  // tile_dim0 (bits 127:112)
  g1[4] = (int)rows;                           // tile_dim1, tile_dim2 = 0
  g1[5] = (int)row_stride;                     // tensor_dim0_stride lo
  g1[6] = 0;                                   // stride hi / dim1_stride lo
  g1[7] = 0;                                   // dim1_stride hi
  v4i z4 = {0, 0, 0, 0};
  v8i z8 = {0, 0, 0, 0, 0, 0, 0, 0};
  __builtin_amdgcn_tensor_load_to_lds(g0, g1, z4, z4, z8, 0);
}

// ---------------------------------------------------------------- convert
__global__ void f2b_kernel(const float* __restrict__ in, u16t* __restrict__ out, int n) {
  int i = blockIdx.x * 256 + threadIdx.x;
  if (i < n) out[i] = f2b(in[i]);
}

// ---------------------------------------------------------------- GEMM
// C[m,n] = sum_k A[m,k] * W[n,k] + bias[n]   (A:[8192,2048] bf16, W:[2048,2048] bf16)
// Block: 128x128 tile, 8 waves (4Mx2N), wave = 32x64, K-step 32.
// Tiles DMA'd by the Tensor Data Mover into double-buffered LDS.
template <bool WRITE_F32>
__global__ __launch_bounds__(256) void gemm_bf16_kernel(
    const u16t* __restrict__ A, const u16t* __restrict__ W,
    const float* __restrict__ bias, float* __restrict__ Cf,
    u16t* __restrict__ Cb) {
  __shared__ u16t At[2][128][40];
  __shared__ u16t Bt[2][128][40];
  const int tid = threadIdx.x;
  const int w = tid >> 5, lane = tid & 31, r = lane & 15, hf = lane >> 4;
  const int wm = w >> 1, wn = w & 1;
  const int m0 = blockIdx.y * 128, n0 = blockIdx.x * 128;
  const u16t* Atile = A + (size_t)m0 * E_;
  const u16t* Wtile = W + (size_t)n0 * E_;
  (void)lane;
  v8f acc[2][4] = {};
  if (w == 0) {                    // wave 0 drives the TDM
    tdm_load_2d(lds_off(&At[0][0][0]), Atile, 32, 128, E_, M_, 3, 3);
    tdm_load_2d(lds_off(&Bt[0][0][0]), Wtile, 32, 128, E_, E_, 3, 3);
  }
  int p = 0;
  for (int k0 = 0; k0 < E_; k0 += 32) {
    if (w == 0) {
      if (k0 + 32 < E_) {          // prefetch next tiles into the other buffer
        tdm_load_2d(lds_off(&At[p ^ 1][0][0]), Atile + k0 + 32, 32, 128, E_, M_, 3, 3);
        tdm_load_2d(lds_off(&Bt[p ^ 1][0][0]), Wtile + k0 + 32, 32, 128, E_, E_, 3, 3);
        __builtin_amdgcn_s_wait_tensorcnt(2);  // wait for current pair only
      } else {
        __builtin_amdgcn_s_wait_tensorcnt(0);
      }
    }
    __syncthreads();               // publish buffer p to all waves
    v16bf af[2], bfr[4];
#pragma unroll
    for (int mi = 0; mi < 2; ++mi) {
      int row = wm * 32 + mi * 16 + r;
      af[mi] = ldfrag(&At[p][row][8 * hf], &At[p][row][16 + 8 * hf]);
    }
#pragma unroll
    for (int ni = 0; ni < 4; ++ni) {
      int row = wn * 64 + ni * 16 + r;
      bfr[ni] = ldfrag(&Bt[p][row][8 * hf], &Bt[p][row][16 + 8 * hf]);
    }
#pragma unroll
    for (int mi = 0; mi < 2; ++mi)
#pragma unroll
      for (int ni = 0; ni < 4; ++ni)
        acc[mi][ni] = wmma_bf16(af[mi], bfr[ni], acc[mi][ni]);
    __syncthreads();               // all reads of buffer p done before reuse
    p ^= 1;
  }
#pragma unroll
  for (int ni = 0; ni < 4; ++ni) {
    int n = n0 + wn * 64 + ni * 16 + r;
    float bv = bias[n];
#pragma unroll
    for (int mi = 0; mi < 2; ++mi) {
#pragma unroll
      for (int v = 0; v < 8; ++v) {
        int m = m0 + wm * 32 + mi * 16 + v + 8 * hf;
        float val = acc[mi][ni][v] + bv;
        size_t idx = (size_t)m * E_ + n;
        if (WRITE_F32) Cf[idx] = val;
        else           Cb[idx] = f2b(val);
      }
    }
  }
}

// ---------------------------------------------------------------- RoPE + permute
// in: bf16 [B,S,H*D]  ->  out: bf16 [B,H,S,D], optional RoPE on (even,odd) pairs
__global__ void rope_perm_kernel(const u16t* __restrict__ in, u16t* __restrict__ out,
                                 int do_rope) {
  int idx = blockIdx.x * 256 + threadIdx.x;     // B*S*H*(D/2) threads exactly
  int i = idx & 63;
  int h = (idx >> 6) & (H_ - 1);
  int s = (idx >> 10) & (S_ - 1);
  int b = idx >> 21;
  const u16t* src = in + (size_t)(b * S_ + s) * E_ + h * D_ + 2 * i;
  float xe = b2f(src[0]), xo = b2f(src[1]);
  if (do_rope) {
    float inv = __expf((-(float)(2 * i) / (float)D_) * 9.210340371976184f); // ln 1e4
    float ang = (float)s * inv;
    float sn, cs;
    __sincosf(ang, &sn, &cs);
    float re = xe * cs - xo * sn;
    float ro = xe * sn + xo * cs;
    xe = re; xo = ro;
  }
  u16t* dst = out + ((size_t)(b * H_ + h) * S_ + s) * D_ + 2 * i;
  dst[0] = f2b(xe);
  dst[1] = f2b(xo);
}

// ---------------------------------------------------------------- flash attention
// Q,K,V: bf16 [B,H,S,D].  O: bf16 [B,S,E] (already permuted back for final GEMM).
// Block: 128 q-rows, 8 waves x 16 rows. K-blocks of 32, causal, online softmax.
// K tile DMA'd by the TDM (overlapped with the VALU V-transpose staging).
__global__ __launch_bounds__(256) void flash_attn_kernel(
    const u16t* __restrict__ Q, const u16t* __restrict__ K,
    const u16t* __restrict__ V, u16t* __restrict__ O) {
  __shared__ u16t Kt[32][136];     // K tile [k][d]  (TDM-padded rows: 128+8 halves)
  __shared__ u16t Vt[128][40];     // V tile transposed [d][k]
  __shared__ u16t Pt[8][16][40];   // per-wave P staging [q][k]
  const int tid = threadIdx.x;
  const int w = tid >> 5, lane = tid & 31, r = lane & 15, hf = lane >> 4;
  const int bh = blockIdx.y;
  const int b = bh >> 4, h = bh & 15;
  const int q0 = blockIdx.x * 128;
  const int qbase = q0 + w * 16;
  const size_t seqbase = (size_t)bh * S_ * D_;
  (void)lane;
  // Q fragments for this wave's 16 rows (4 d-chunks of 32)
  v16bf qf[4];
  {
    const u16t* qrow = Q + seqbase + (size_t)(qbase + r) * D_;
#pragma unroll
    for (int c = 0; c < 4; ++c)
      qf[c] = ldfrag(qrow + 32 * c + 8 * hf, qrow + 32 * c + 16 + 8 * hf);
  }
  v8f oacc[8] = {};
  float mrow[8], lrow[8];
#pragma unroll
  for (int v = 0; v < 8; ++v) { mrow[v] = -1e30f; lrow[v] = 0.0f; }
  const float scale = 0.0883883476483184f;  // 1/sqrt(128)
  const int kend = q0 + 128;
  for (int k0 = 0; k0 < kend; k0 += 32) {
    __syncthreads();
    if (w == 0)   // TDM: K tile 32 rows x 128 halves, pad 256B rows by 16B -> 136 stride
      tdm_load_2d(lds_off(&Kt[0][0]), K + seqbase + (size_t)k0 * D_,
                  128, 32, D_, S_, 5, 3);
    {   // cooperative transposed V staging (TDM cannot transpose)
      int row = tid >> 3, seg = (tid & 7) * 16;
      const u16t* vs = V + seqbase + (size_t)(k0 + row) * D_ + seg;
#pragma unroll
      for (int j = 0; j < 16; ++j)
        Vt[seg + j][row] = vs[j];
    }
    if (w == 0) __builtin_amdgcn_s_wait_tensorcnt(0);
    __syncthreads();
    if (k0 > qbase + 15) continue;   // fully masked for this wave; barriers stay uniform
    // S = Q K^T  (16 x 32 scores = 2 N-tiles x 4 d-chunks)
    v8f sacc[2] = {};
#pragma unroll
    for (int t = 0; t < 2; ++t)
#pragma unroll
      for (int c = 0; c < 4; ++c) {
        v16bf kf = ldfrag(&Kt[16 * t + r][32 * c + 8 * hf],
                          &Kt[16 * t + r][32 * c + 16 + 8 * hf]);
        sacc[t] = wmma_bf16(qf[c], kf, sacc[t]);
      }
    // online softmax; row (q = qbase+v+8*hf) lives across 16 lanes of one half
#pragma unroll
    for (int v = 0; v < 8; ++v) {
      int q = qbase + v + 8 * hf;
      float s0 = sacc[0][v] * scale;
      float s1 = sacc[1][v] * scale;
      if (k0 + r > q)      s0 = -1e30f;
      if (k0 + 16 + r > q) s1 = -1e30f;
      float mx = fmaxf(s0, s1);
      mx = fmaxf(mx, __shfl_xor(mx, 1, 32));
      mx = fmaxf(mx, __shfl_xor(mx, 2, 32));
      mx = fmaxf(mx, __shfl_xor(mx, 4, 32));
      mx = fmaxf(mx, __shfl_xor(mx, 8, 32));
      float mn = fmaxf(mrow[v], mx);
      float corr = __expf(mrow[v] - mn);
      mrow[v] = mn;
      float p0 = __expf(s0 - mn);
      float p1 = __expf(s1 - mn);
      float rs = p0 + p1;
      rs += __shfl_xor(rs, 1, 32);
      rs += __shfl_xor(rs, 2, 32);
      rs += __shfl_xor(rs, 4, 32);
      rs += __shfl_xor(rs, 8, 32);
      lrow[v] = lrow[v] * corr + rs;
#pragma unroll
      for (int t = 0; t < 8; ++t) oacc[t][v] *= corr;
      Pt[w][v + 8 * hf][r]      = f2b(p0);
      Pt[w][v + 8 * hf][16 + r] = f2b(p1);
    }
    // O += P V   (A-frag of P from LDS, B-frags of V^T from LDS)
    v16bf pf = ldfrag(&Pt[w][r][8 * hf], &Pt[w][r][16 + 8 * hf]);
#pragma unroll
    for (int t = 0; t < 8; ++t) {
      v16bf vf = ldfrag(&Vt[16 * t + r][8 * hf], &Vt[16 * t + r][16 + 8 * hf]);
      oacc[t] = wmma_bf16(pf, vf, oacc[t]);
    }
  }
  // write O[b][q][h*D + d] in bf16 (input to output projection)
#pragma unroll
  for (int v = 0; v < 8; ++v) {
    float inv_l = 1.0f / lrow[v];
    int q = qbase + v + 8 * hf;
    u16t* orow = O + (size_t)(b * S_ + q) * E_ + h * D_;
#pragma unroll
    for (int t = 0; t < 8; ++t)
      orow[16 * t + r] = f2b(oacc[t][v] * inv_l);
  }
}

// ---------------------------------------------------------------- launcher
extern "C" void kernel_launch(void* const* d_in, const int* in_sizes, int n_in,
                              void* d_out, int out_size, void* d_ws, size_t ws_size,
                              hipStream_t stream) {
  (void)in_sizes; (void)n_in; (void)out_size; (void)ws_size;
  const float* x  = (const float*)d_in[0];
  const float* Wq = (const float*)d_in[1];
  const float* bq = (const float*)d_in[2];
  const float* Wk = (const float*)d_in[3];
  const float* bk = (const float*)d_in[4];
  const float* Wv = (const float*)d_in[5];
  const float* bv = (const float*)d_in[6];
  const float* Wo = (const float*)d_in[7];
  const float* bo = (const float*)d_in[8];
  float* out = (float*)d_out;

  u16t* ws = (u16t*)d_ws;
  const size_t NX = (size_t)M_ * E_;   // 16,777,216 elements
  const size_t NW = (size_t)E_ * E_;   //  4,194,304 elements
  u16t* xb  = ws;            // x bf16                 (later reused as Qb)
  u16t* wqb = xb + NX;
  u16t* wkb = wqb + NW;
  u16t* wvb = wkb + NW;
  u16t* wob = wvb + NW;
  u16t* qp  = wob + NW;      // Q proj [B,S,E]         (later reused as Kb)
  u16t* kp  = qp + NX;       // K proj                 (later reused as Vb)
  u16t* vp  = kp + NX;       // V proj                 (later reused as attn out)
  u16t* qb  = xb;            // RoPE'd Q [B,H,S,D]
  u16t* kb  = qp;            // RoPE'd K [B,H,S,D]
  u16t* vb  = kp;            // V       [B,H,S,D]
  u16t* ob  = vp;            // attention out [B,S,E]

  dim3 blk(256);
  f2b_kernel<<<(unsigned)((NX + 255) / 256), blk, 0, stream>>>(x,  xb,  (int)NX);
  f2b_kernel<<<(unsigned)((NW + 255) / 256), blk, 0, stream>>>(Wq, wqb, (int)NW);
  f2b_kernel<<<(unsigned)((NW + 255) / 256), blk, 0, stream>>>(Wk, wkb, (int)NW);
  f2b_kernel<<<(unsigned)((NW + 255) / 256), blk, 0, stream>>>(Wv, wvb, (int)NW);
  f2b_kernel<<<(unsigned)((NW + 255) / 256), blk, 0, stream>>>(Wo, wob, (int)NW);

  dim3 ggrid(E_ / 128, M_ / 128);
  gemm_bf16_kernel<false><<<ggrid, blk, 0, stream>>>(xb, wqb, bq, nullptr, qp);
  gemm_bf16_kernel<false><<<ggrid, blk, 0, stream>>>(xb, wkb, bk, nullptr, kp);
  gemm_bf16_kernel<false><<<ggrid, blk, 0, stream>>>(xb, wvb, bv, nullptr, vp);

  const int rthreads = B_ * S_ * H_ * (D_ / 2);  // 8,388,608
  rope_perm_kernel<<<rthreads / 256, blk, 0, stream>>>(qp, qb, 1);
  rope_perm_kernel<<<rthreads / 256, blk, 0, stream>>>(kp, kb, 1);
  rope_perm_kernel<<<rthreads / 256, blk, 0, stream>>>(vp, vb, 0);

  dim3 agrid(S_ / 128, B_ * H_);
  flash_attn_kernel<<<agrid, blk, 0, stream>>>(qb, kb, vb, ob);

  gemm_bf16_kernel<true><<<ggrid, blk, 0, stream>>>(ob, wob, bo, out, nullptr);
}
